// Model_77481210020616
// MI455X (gfx1250) — matrix operand order repaired and checked by reading, not compiled, not run on hardware
//
#include <hip/hip_runtime.h>
#include <hip/hip_bf16.h>

// ---------------- problem constants (from reference) ----------------
#define BB 32
#define SS 512
#define TT 128
#define VV 128
#define LL 5
#define EE 512
#define HH 1024
#define FF 2048
#define START_CHAR 1
#define SEP_LANG (LL - 1)

typedef __attribute__((ext_vector_type(16))) __bf16 v16bf;
typedef __attribute__((ext_vector_type(8)))  __bf16 v8bf;
typedef __attribute__((ext_vector_type(8)))  float  fvec8;

#define FLAG_ACC  1
#define FLAG_RELU 2

// ---------------------------------------------------------------
// f32 -> bf16 convert (weights go L2-resident at half footprint)
// ---------------------------------------------------------------
__global__ void k_f32_to_bf16(const float* __restrict__ src,
                              __bf16* __restrict__ dst, long n) {
    long i = (long)blockIdx.x * blockDim.x + threadIdx.x;
    if (i < n) dst[i] = (__bf16)src[i];
}

// ---------------------------------------------------------------
// gather char/lang embeddings into a (ntok, 2E) bf16 matrix
// chars/langs may be null -> use constant index
// ---------------------------------------------------------------
__global__ void k_gather(const int* __restrict__ chars, const int* __restrict__ langs,
                         int const_char, int const_lang,
                         const float* __restrict__ ce, const float* __restrict__ le,
                         __bf16* __restrict__ X, long ntok) {
    long idx = (long)blockIdx.x * blockDim.x + threadIdx.x;
    if (idx >= ntok * (2 * EE)) return;
    long t = idx >> 10;          // / 1024
    int  e = (int)(idx & 1023);
    if (e < EE) {
        int c = chars ? chars[t] : const_char;
        X[idx] = (__bf16)ce[(long)c * EE + e];
    } else {
        int l = langs ? langs[t] : const_lang;
        X[idx] = (__bf16)le[(long)l * EE + (e - EE)];
    }
}

// ---------------------------------------------------------------
// Generic WMMA GEMM:  C[M,N] = act( A[M,K](bf16,row-major,lda)
//                                  @ W[N,K](bf16,row-major,ldw)^T + bias )
// One wave computes a 32x16 C tile: TWO v_wmma_f32_16x16x32_bf16 per
// K-step sharing a single B fragment (halves weight traffic -- these
// skinny M=32 GEMMs are L2-weight-bandwidth bound on the serial chain).
// A fragment: lane(0-15)=row, two 16B loads (K 0-7 / 16-23); lanes 16-31
//             get K 8-15 / 24-31 (ISA 16-bit A layout).
// B fragment: lane = output column, two 16B loads along W's row
//             (ISA 16-bit B layout: lanes 0-15 K 0-7/16-23, lanes 16-31 K 8-15/24-31).
// ---------------------------------------------------------------
__global__ __launch_bounds__(128)
void k_gemm_bf16_wmma(const __bf16* __restrict__ A, int lda,
                      const __bf16* __restrict__ W, int ldw,
                      const float* __restrict__ bias,
                      float* __restrict__ Cf, __bf16* __restrict__ Cbf,
                      int ldc, int M, int N, int K, int flags) {
    const int wave   = threadIdx.x >> 5;
    const int lane   = threadIdx.x & 31;
    const int half   = lane >> 4;       // 0: lanes 0-15, 1: lanes 16-31
    const int lane16 = lane & 15;

    const int  tiles_n = N >> 4;
    const long tile    = (long)blockIdx.x * 4 + wave;
    const long total   = (long)(M >> 5) * tiles_n;   // 32-row m-tiles
    if (tile >= total) return;
    const int tn = (int)(tile % tiles_n);
    const int tm = (int)(tile / tiles_n);
    const int m0 = tm << 5, n0 = tn << 4;

    const __bf16* Arow0 = A + (long)(m0 + lane16) * lda;        // rows m0..m0+15
    const __bf16* Arow1 = A + (long)(m0 + 16 + lane16) * lda;   // rows m0+16..m0+31
    const __bf16* Wrow  = W + (long)(n0 + lane16) * ldw;

    fvec8 acc0 = {}, acc1 = {};
    if (flags & FLAG_ACC) {
#pragma unroll
        for (int r = 0; r < 8; ++r) {
            acc0[r] = Cf[(long)(m0 + r + half * 8) * ldc + (n0 + lane16)];
            acc1[r] = Cf[(long)(m0 + 16 + r + half * 8) * ldc + (n0 + lane16)];
        }
    }

    for (int k0 = 0; k0 < K; k0 += 32) {
        // prefetch a few iterations ahead (global_prefetch_b8)
        __builtin_prefetch((const void*)(Wrow + k0 + 256), 0, 1);
        __builtin_prefetch((const void*)(Arow0 + k0 + 256), 0, 1);
        __builtin_prefetch((const void*)(Arow1 + k0 + 256), 0, 1);

        v8bf blo = *(const v8bf*)(Wrow + k0 + half * 8);        // K: 0-7  / 8-15
        v8bf bhi = *(const v8bf*)(Wrow + k0 + 16 + half * 8);   // K: 16-23/ 24-31
        v16bf b;
#pragma unroll
        for (int i = 0; i < 8; ++i) { b[i] = blo[i]; b[i + 8] = bhi[i]; }

        v8bf a0lo = *(const v8bf*)(Arow0 + k0 + half * 8);      // K: 0-7  / 8-15
        v8bf a0hi = *(const v8bf*)(Arow0 + k0 + 16 + half * 8); // K: 16-23/ 24-31
        v8bf a1lo = *(const v8bf*)(Arow1 + k0 + half * 8);
        v8bf a1hi = *(const v8bf*)(Arow1 + k0 + 16 + half * 8);
        v16bf a0, a1;
#pragma unroll
        for (int i = 0; i < 8; ++i) {
            a0[i] = a0lo[i]; a0[i + 8] = a0hi[i];
            a1[i] = a1lo[i]; a1[i + 8] = a1hi[i];
        }

        acc0 = __builtin_amdgcn_wmma_f32_16x16x32_bf16(
            false, a0, false, b, (short)0, acc0, false, false);
        acc1 = __builtin_amdgcn_wmma_f32_16x16x32_bf16(
            false, a1, false, b, (short)0, acc1, false, false);
    }

    const int col = n0 + lane16;
    const float bv = bias ? bias[col] : 0.0f;
#pragma unroll
    for (int r = 0; r < 8; ++r) {
        const int row0 = m0 + r + half * 8;
        const int row1 = row0 + 16;
        float v0 = acc0[r] + bv;
        float v1 = acc1[r] + bv;
        if (flags & FLAG_RELU) { v0 = fmaxf(v0, 0.0f); v1 = fmaxf(v1, 0.0f); }
        if (Cf)  { Cf[(long)row0 * ldc + col] = v0;
                   Cf[(long)row1 * ldc + col] = v1; }
        if (Cbf) { Cbf[(long)row0 * ldc + col] = (__bf16)v0;
                   Cbf[(long)row1 * ldc + col] = (__bf16)v1; }
    }
}

// ---------------------------------------------------------------
// GRU gate nonlinearity.  gi = x@Wi^T (bf16, no bias yet),
// gh = h@Wh^T (bf16, optional; null => zeros i.e. h_prev==0 path).
// h = (1-z)*tanh(in + r*hn) + z*h_prev,  r/z = sigmoid.
// Optionally scatter h (bf16) into enc_states[:, s, :].
// ---------------------------------------------------------------
__global__ void k_gru_gates(const __bf16* __restrict__ gi, long gi_stride,
                            const __bf16* __restrict__ gh,
                            const float* __restrict__ bi, const float* __restrict__ bh,
                            const float* __restrict__ h_prev,
                            float* __restrict__ h_out, __bf16* __restrict__ h_out_bf,
                            __bf16* __restrict__ state_out, long state_stride) {
    int idx = blockIdx.x * blockDim.x + threadIdx.x;
    if (idx >= BB * HH) return;
    int b = idx / HH, j = idx % HH;

    const __bf16* gib = gi + (long)b * gi_stride;
    float ir  = (float)gib[j]          + bi[j];
    float iz  = (float)gib[j + HH]     + bi[j + HH];
    float inn = (float)gib[j + 2 * HH] + bi[j + 2 * HH];
    float hr = bh[j], hz = bh[j + HH], hn = bh[j + 2 * HH];
    if (gh) {
        const __bf16* ghb = gh + (long)b * 3 * HH;
        hr += (float)ghb[j];
        hz += (float)ghb[j + HH];
        hn += (float)ghb[j + 2 * HH];
    }
    float hp = h_prev ? h_prev[idx] : 0.0f;
    float r = 1.0f / (1.0f + __expf(-(ir + hr)));
    float z = 1.0f / (1.0f + __expf(-(iz + hz)));
    float n = tanhf(inn + r * hn);
    float h = (1.0f - z) * n + z * hp;
    h_out[idx] = h;
    h_out_bf[idx] = (__bf16)h;
    if (state_out) state_out[(long)b * state_stride + j] = (__bf16)h;
}

// mlp_in = h + ctx, emitted as bf16 GEMM operand
__global__ void k_add_to_bf(const float* __restrict__ a, const float* __restrict__ b,
                            __bf16* __restrict__ out, int n) {
    int i = blockIdx.x * blockDim.x + threadIdx.x;
    if (i < n) out[i] = (__bf16)(a[i] + b[i]);
}

// ctx0 = enc_states[:, S-1, :]
__global__ void k_ctx0(const __bf16* __restrict__ enc, float* __restrict__ ctx,
                       __bf16* __restrict__ ctx_bf) {
    int idx = blockIdx.x * blockDim.x + threadIdx.x;
    if (idx >= BB * HH) return;
    int b = idx / HH, j = idx % HH;
    float v = (float)enc[((long)b * SS + (SS - 1)) * HH + j];
    ctx[idx] = v;
    ctx_bf[idx] = (__bf16)v;
}

// out[b, :] = [ xa[b, 0:na] , xb[b, 0:nb] ]   (bf16)
__global__ void k_concat(const __bf16* __restrict__ xa, long lda, int na,
                         const __bf16* __restrict__ xb, long ldb, int nb,
                         __bf16* __restrict__ out) {
    int w = na + nb;
    int idx = blockIdx.x * blockDim.x + threadIdx.x;
    if (idx >= BB * w) return;
    int b = idx / w, j = idx % w;
    out[idx] = (j < na) ? xa[(long)b * lda + j] : xb[(long)b * ldb + (j - na)];
}

// att[b,s] = enc_states[b,s,:] . q[b,:]   (one wave per (b,s))
__global__ void k_attn_dot(const __bf16* __restrict__ enc, const __bf16* __restrict__ q,
                           float* __restrict__ att) {
    int lane = threadIdx.x & 31;
    long w = (long)blockIdx.x * 4 + (threadIdx.x >> 5);
    if (w >= (long)BB * SS) return;
    int b = (int)(w / SS), s = (int)(w % SS);
    const v8bf* ev = (const v8bf*)(enc + ((long)b * SS + s) * HH);
    const v8bf* qv = (const v8bf*)(q + (long)b * HH);
    float sum = 0.0f;
    for (int i = lane; i < HH / 8; i += 32) {
        v8bf a = ev[i], c = qv[i];
#pragma unroll
        for (int j = 0; j < 8; ++j) sum += (float)a[j] * (float)c[j];
    }
#pragma unroll
    for (int o = 16; o > 0; o >>= 1) sum += __shfl_xor(sum, o, 32);
    if (lane == 0) att[(long)b * SS + s] = sum;
}

// per-batch softmax over S
__global__ void k_softmax(const float* __restrict__ att, float* __restrict__ w) {
    __shared__ float red[256];
    int b = blockIdx.x, tid = threadIdx.x;
    const float* a = att + (long)b * SS;
    float* wb = w + (long)b * SS;
    float m = -3.4e38f;
    for (int i = tid; i < SS; i += 256) m = fmaxf(m, a[i]);
    red[tid] = m; __syncthreads();
    for (int o = 128; o > 0; o >>= 1) { if (tid < o) red[tid] = fmaxf(red[tid], red[tid + o]); __syncthreads(); }
    float mx = red[0]; __syncthreads();
    float s = 0.0f;
    for (int i = tid; i < SS; i += 256) { float e = __expf(a[i] - mx); wb[i] = e; s += e; }
    red[tid] = s; __syncthreads();
    for (int o = 128; o > 0; o >>= 1) { if (tid < o) red[tid] += red[tid + o]; __syncthreads(); }
    float inv = 1.0f / red[0]; __syncthreads();
    for (int i = tid; i < SS; i += 256) wb[i] *= inv;
}

// ctx[b,h] = sum_s w[b,s] * att_vals[b,s,h]
__global__ void k_context(const float* __restrict__ w, const __bf16* __restrict__ av,
                          float* __restrict__ ctx, __bf16* __restrict__ ctx_bf) {
    int idx = blockIdx.x * blockDim.x + threadIdx.x;
    if (idx >= BB * HH) return;
    int b = idx / HH, h = idx % HH;
    const float* wb = w + (long)b * SS;
    const __bf16* a = av + (long)b * SS * HH + h;
    float s = 0.0f;
    for (int si = 0; si < SS; ++si) s += wb[si] * (float)a[(long)si * HH];
    ctx[idx] = s;
    ctx_bf[idx] = (__bf16)s;
}

// ======================= host orchestration =======================
extern "C" void kernel_launch(void* const* d_in, const int* in_sizes, int n_in,
                              void* d_out, int out_size, void* d_ws, size_t ws_size,
                              hipStream_t stream) {
    const int* src_chars = (const int*)d_in[0];
    const int* src_langs = (const int*)d_in[1];
    const int* tgt_chars = (const int*)d_in[2];
    const float* char_emb = (const float*)d_in[3];
    const float* lang_emb = (const float*)d_in[4];
    const float* fc_W   = (const float*)d_in[5];
    const float* fc_b   = (const float*)d_in[6];
    const float* enc_Wi = (const float*)d_in[7];
    const float* enc_Wh = (const float*)d_in[8];
    const float* enc_bi = (const float*)d_in[9];
    const float* enc_bh = (const float*)d_in[10];
    const float* dec_Wi = (const float*)d_in[11];
    const float* dec_Wh = (const float*)d_in[12];
    const float* dec_bi = (const float*)d_in[13];
    const float* dec_bh = (const float*)d_in[14];
    const float* Wq  = (const float*)d_in[15];
    const float* Wk  = (const float*)d_in[16];
    const float* Wcs = (const float*)d_in[17];
    const float* W1  = (const float*)d_in[18];
    const float* b1  = (const float*)d_in[19];
    const float* W2  = (const float*)d_in[20];
    const float* b2  = (const float*)d_in[21];
    const float* W3  = (const float*)d_in[22];
    float* out = (float*)d_out;

    // ---- workspace allocator (256B aligned) ----
    char* wp = (char*)d_ws;
    auto alloc = [&](size_t bytes) -> void* {
        void* r = (void*)wp;
        wp += (bytes + 255) & ~(size_t)255;
        return r;
    };
    auto bfbuf = [&](long elems) { return (__bf16*)alloc((size_t)elems * 2); };
    auto f32buf = [&](long elems) { return (float*)alloc((size_t)elems * 4); };

    // bf16 weights (L2-resident working set ~54MB)
    __bf16* fcW_bf   = bfbuf((long)EE * 2 * EE);
    __bf16* encWi_bf = bfbuf((long)3 * HH * EE);
    __bf16* encWh_bf = bfbuf((long)3 * HH * HH);
    __bf16* decWi_bf = bfbuf((long)3 * HH * (EE + HH));
    __bf16* decWh_bf = bfbuf((long)3 * HH * HH);
    __bf16* Wq_bf    = bfbuf((long)HH * HH);
    __bf16* Wk_bf    = bfbuf((long)HH * HH);
    __bf16* Wcs_bf   = bfbuf((long)HH * EE);
    __bf16* W1_bf    = bfbuf((long)2 * FF * HH);
    __bf16* W2_bf    = bfbuf((long)FF * 2 * FF);
    __bf16* W3_bf    = bfbuf((long)VV * FF);

    // activations
    const long BS = (long)BB * SS, BT = (long)BB * TT;
    __bf16* Xsrc       = bfbuf(BS * 2 * EE);
    __bf16* encoded_bf = bfbuf(BS * EE);
    __bf16* gienc_bf   = bfbuf(BS * 3 * HH);
    __bf16* encsts_bf  = bfbuf(BS * HH);
    float*  attv_f32   = f32buf(BS * HH);
    __bf16* attv_bf    = bfbuf(BS * HH);
    __bf16* Xtgt       = bfbuf(BT * 2 * EE);
    __bf16* tgtfc_bf   = bfbuf(BT * EE);
    __bf16* Xstart     = bfbuf((long)BB * 2 * EE);
    __bf16* startfc_bf = bfbuf((long)BB * EE);
    float*  h_f32      = f32buf((long)BB * HH);
    __bf16* h_bf       = bfbuf((long)BB * HH);
    float*  ctx_f32    = f32buf((long)BB * HH);
    __bf16* ctx_bf     = bfbuf((long)BB * HH);
    __bf16* decin_bf   = bfbuf((long)BB * (EE + HH));
    __bf16* gi_bf      = bfbuf((long)BB * 3 * HH);
    __bf16* gh_bf      = bfbuf((long)BB * 3 * HH);
    __bf16* mlpin_bf   = bfbuf((long)BB * HH);
    __bf16* h1_bf      = bfbuf((long)BB * 2 * FF);
    __bf16* h2_bf      = bfbuf((long)BB * FF);
    __bf16* q_bf       = bfbuf((long)BB * HH);
    float*  att_f32    = f32buf((long)BB * SS);
    float*  wsm_f32    = f32buf((long)BB * SS);

    auto cvt = [&](const float* s, __bf16* d, long n) {
        k_f32_to_bf16<<<(int)((n + 255) / 256), 256, 0, stream>>>(s, d, n);
    };
    auto gemm = [&](const __bf16* A, int lda, const __bf16* W, int ldw,
                    const float* bias, float* Cf, __bf16* Cbf, int ldc,
                    int M, int N, int K, int flags) {
        long tiles = (long)(M / 32) * (N / 16);   // 32x16 C tile per wave
        k_gemm_bf16_wmma<<<(int)((tiles + 3) / 4), 128, 0, stream>>>(
            A, lda, W, ldw, bias, Cf, Cbf, ldc, M, N, K, flags);
    };

    // 1) weights -> bf16
    cvt(fc_W, fcW_bf, (long)EE * 2 * EE);
    cvt(enc_Wi, encWi_bf, (long)3 * HH * EE);
    cvt(enc_Wh, encWh_bf, (long)3 * HH * HH);
    cvt(dec_Wi, decWi_bf, (long)3 * HH * (EE + HH));
    cvt(dec_Wh, decWh_bf, (long)3 * HH * HH);
    cvt(Wq, Wq_bf, (long)HH * HH);
    cvt(Wk, Wk_bf, (long)HH * HH);
    cvt(Wcs, Wcs_bf, (long)HH * EE);
    cvt(W1, W1_bf, (long)2 * FF * HH);
    cvt(W2, W2_bf, (long)FF * 2 * FF);
    cvt(W3, W3_bf, (long)VV * FF);

    // 2) embedding gathers
    k_gather<<<(int)((BS * 2 * EE + 255) / 256), 256, 0, stream>>>(
        src_chars, src_langs, 0, 0, char_emb, lang_emb, Xsrc, BS);
    k_gather<<<(int)((BT * 2 * EE + 255) / 256), 256, 0, stream>>>(
        tgt_chars, nullptr, 0, SEP_LANG, char_emb, lang_emb, Xtgt, BT);
    k_gather<<<(int)(((long)BB * 2 * EE + 255) / 256), 256, 0, stream>>>(
        nullptr, nullptr, START_CHAR, SEP_LANG, char_emb, lang_emb, Xstart, BB);

    // 3) fc projections (encoded, tgt_emb, start_emb)
    gemm(Xsrc, 2 * EE, fcW_bf, 2 * EE, fc_b, nullptr, encoded_bf, EE, (int)BS, EE, 2 * EE, 0);
    gemm(Xtgt, 2 * EE, fcW_bf, 2 * EE, fc_b, nullptr, tgtfc_bf, EE, (int)BT, EE, 2 * EE, 0);
    gemm(Xstart, 2 * EE, fcW_bf, 2 * EE, fc_b, nullptr, startfc_bf, EE, BB, EE, 2 * EE, 0);

    // 4) encoder input gates for ALL timesteps in one big GEMM
    gemm(encoded_bf, EE, encWi_bf, EE, nullptr, nullptr, gienc_bf, 3 * HH, (int)BS, 3 * HH, EE, 0);

    // 5) encoder recurrence (serial over S; only h@Wh is on the chain)
    hipMemsetAsync(h_f32, 0, (size_t)BB * HH * 4, stream);
    hipMemsetAsync(h_bf, 0, (size_t)BB * HH * 2, stream);
    const int gate_blocks = (BB * HH + 255) / 256;
    for (int s = 0; s < SS; ++s) {
        gemm(h_bf, HH, encWh_bf, HH, nullptr, nullptr, gh_bf, 3 * HH, BB, 3 * HH, HH, 0);
        k_gru_gates<<<gate_blocks, 256, 0, stream>>>(
            gienc_bf + (long)s * 3 * HH, (long)SS * 3 * HH, gh_bf,
            enc_bi, enc_bh, h_f32, h_f32, h_bf,
            encsts_bf + (long)s * HH, (long)SS * HH);
    }

    // 6) att_vals = encoded@Wcs^T + enc_states@Wk^T
    gemm(encoded_bf, EE, Wcs_bf, EE, nullptr, attv_f32, nullptr, HH, (int)BS, HH, EE, 0);
    gemm(encsts_bf, HH, Wk_bf, HH, nullptr, attv_f32, attv_bf, HH, (int)BS, HH, HH, FLAG_ACC);

    // 7) decoder init: ctx0 = enc_states[:,-1]; h from [start_emb, ctx0], h_prev=0
    k_ctx0<<<gate_blocks, 256, 0, stream>>>(encsts_bf, ctx_f32, ctx_bf);
    k_concat<<<(BB * (EE + HH) + 255) / 256, 256, 0, stream>>>(
        startfc_bf, EE, EE, ctx_bf, HH, HH, decin_bf);
    gemm(decin_bf, EE + HH, decWi_bf, EE + HH, nullptr, nullptr, gi_bf, 3 * HH, BB, 3 * HH, EE + HH, 0);
    hipMemsetAsync(h_f32, 0, (size_t)BB * HH * 4, stream);
    k_gru_gates<<<gate_blocks, 256, 0, stream>>>(
        gi_bf, 3 * HH, nullptr, dec_bi, dec_bh, h_f32, h_f32, h_bf, nullptr, 0);

    // 8) decoder loop
    for (int t = 0; t < TT; ++t) {
        // scores_t = MLP(h + ctx)   -- dominant WMMA work
        k_add_to_bf<<<gate_blocks, 256, 0, stream>>>(h_f32, ctx_f32, mlpin_bf, BB * HH);
        gemm(mlpin_bf, HH, W1_bf, HH, b1, nullptr, h1_bf, 2 * FF, BB, 2 * FF, HH, FLAG_RELU);
        gemm(h1_bf, 2 * FF, W2_bf, 2 * FF, b2, nullptr, h2_bf, FF, BB, FF, 2 * FF, FLAG_RELU);
        gemm(h2_bf, FF, W3_bf, FF, nullptr, out + (long)t * VV, nullptr, TT * VV, BB, VV, FF, 0);

        // attention
        gemm(h_bf, HH, Wq_bf, HH, nullptr, nullptr, q_bf, HH, BB, HH, HH, 0);
        k_attn_dot<<<(int)((BS + 3) / 4), 128, 0, stream>>>(encsts_bf, q_bf, att_f32);
        k_softmax<<<BB, 256, 0, stream>>>(att_f32, wsm_f32);
        k_context<<<gate_blocks, 256, 0, stream>>>(wsm_f32, attv_bf, ctx_f32, ctx_bf);

        // GRU update with [tgt_emb_t, ctx_new]
        k_concat<<<(BB * (EE + HH) + 255) / 256, 256, 0, stream>>>(
            tgtfc_bf + (long)t * EE, (long)TT * EE, EE, ctx_bf, HH, HH, decin_bf);
        gemm(decin_bf, EE + HH, decWi_bf, EE + HH, nullptr, nullptr, gi_bf, 3 * HH, BB, 3 * HH, EE + HH, 0);
        gemm(h_bf, HH, decWh_bf, HH, nullptr, nullptr, gh_bf, 3 * HH, BB, 3 * HH, HH, 0);
        k_gru_gates<<<gate_blocks, 256, 0, stream>>>(
            gi_bf, 3 * HH, gh_bf, dec_bi, dec_bh, h_f32, h_f32, h_bf, nullptr, 0);
    }
}
